// Net_16673063043121
// MI455X (gfx1250) — compile-verified, matrix operand order checked by reading, not run on hardware
//
#include <hip/hip_runtime.h>

#define N_NODES 100000
#define N_EDGES 3200000
#define F_IN    512
#define HID     512
#define N_CLS   64
#define K_STEPS 10
#define ALPHA   0.1f

typedef __attribute__((ext_vector_type(16))) _Float16 v16h;
typedef __attribute__((ext_vector_type(8)))  _Float16 v8h;
typedef __attribute__((ext_vector_type(4)))  _Float16 v4h;
typedef __attribute__((ext_vector_type(8)))  float    v8f;

__device__ __forceinline__ v16h cat8(v8h lo, v8h hi) {
    return __builtin_shufflevector(lo, hi, 0,1,2,3,4,5,6,7,8,9,10,11,12,13,14,15);
}

// ---------------------------------------------------------------------------
// f32 -> f16 conversion (vectorized 4-wide)
// ---------------------------------------------------------------------------
__global__ void cvt_f32_f16_v4(const float* __restrict__ in, _Float16* __restrict__ out, long n4) {
    long i = (long)blockIdx.x * blockDim.x + threadIdx.x;
    if (i >= n4) return;
    float4 v = ((const float4*)in)[i];
    v4h h;
    h.x = (_Float16)v.x; h.y = (_Float16)v.y; h.z = (_Float16)v.z; h.w = (_Float16)v.w;
    ((v4h*)out)[i] = h;
}

// in: [R, C] f32 row-major  ->  out: [C, R] f16 (transposed)
__global__ void cvt_transpose_f16(const float* __restrict__ in, _Float16* __restrict__ out,
                                  int R, int C) {
    int idx = blockIdx.x * blockDim.x + threadIdx.x;
    if (idx >= R * C) return;
    int r = idx / C, c = idx % C;
    out[(size_t)c * R + r] = (_Float16)in[idx];
}

// ---------------------------------------------------------------------------
// WMMA GEMM: D = A[MxK] * Bt[NxK]^T + bias.  Wave computes a 16x32 tile
// (2 accumulators) -- this register footprint lets the compiler unroll K by 8
// and keep loads >=2 iterations in flight across each v_wmma (observed:
// s_wait_loadcnt 0x2/0x1 steady state).  Block = 128 threads (4 waves).
// ---------------------------------------------------------------------------
__device__ __forceinline__ void wmma_tile(const _Float16* __restrict__ A,
                                          const _Float16* __restrict__ Bt,
                                          int K, int row, int n0, int n1, int khi,
                                          v8f& c0, v8f& c1) {
    const _Float16* ap  = A  + (size_t)row * K + khi * 8;
    const _Float16* b0p = Bt + (size_t)n0  * K + khi * 8;
    const _Float16* b1p = Bt + (size_t)n1  * K + khi * 8;
    for (int kb = 0; kb < K; kb += 32) {
        v16h a  = cat8(*(const v8h*)(ap  + kb), *(const v8h*)(ap  + kb + 16));
        v16h b0 = cat8(*(const v8h*)(b0p + kb), *(const v8h*)(b0p + kb + 16));
        v16h b1 = cat8(*(const v8h*)(b1p + kb), *(const v8h*)(b1p + kb + 16));
        c0 = __builtin_amdgcn_wmma_f32_16x16x32_f16(false, a, false, b0, (short)0, c0, false, false);
        c1 = __builtin_amdgcn_wmma_f32_16x16x32_f16(false, a, false, b1, (short)0, c1, false, false);
    }
}

// GEMM1: out = relu(A*B + bias), stored as f16 [M, N]
__global__ void gemm_wmma_relu_f16(const _Float16* __restrict__ A,
                                   const _Float16* __restrict__ Bt,
                                   const float* __restrict__ bias,
                                   _Float16* __restrict__ Out,
                                   int M, int N, int K) {
    const int lane  = threadIdx.x & 31;
    const int wave  = threadIdx.x >> 5;
    const int mbase = blockIdx.y * 64 + wave * 16;
    if (mbase >= M) return;                       // wave-uniform guard, EXEC stays full
    const int nbase = blockIdx.x * 32;
    const int lrow  = lane & 15;
    const int khi   = lane >> 4;
    const int row   = mbase + lrow;
    const int n0    = nbase + lrow;
    const int n1    = n0 + 16;

    v8f c0 = {}; v8f c1 = {};
    wmma_tile(A, Bt, K, row, n0, n1, khi, c0, c1);

    const float bv0 = bias[n0], bv1 = bias[n1];
    #pragma unroll
    for (int r = 0; r < 8; ++r) {
        int orow = mbase + 8 * khi + r;           // D layout: VGPR r -> M = r + 8*(l>>4)
        float v0 = c0[r] + bv0; v0 = v0 > 0.f ? v0 : 0.f;
        float v1 = c1[r] + bv1; v1 = v1 > 0.f ? v1 : 0.f;
        Out[(size_t)orow * N + n0] = (_Float16)v0;
        Out[(size_t)orow * N + n1] = (_Float16)v1;
    }
}

// GEMM2: out = A*B + bias, stored as f32 into two buffers (h0 and initial h)
__global__ void gemm_wmma_f32x2(const _Float16* __restrict__ A,
                                const _Float16* __restrict__ Bt,
                                const float* __restrict__ bias,
                                float* __restrict__ Out0,
                                float* __restrict__ Out1,
                                int M, int N, int K) {
    const int lane  = threadIdx.x & 31;
    const int wave  = threadIdx.x >> 5;
    const int mbase = blockIdx.y * 64 + wave * 16;
    if (mbase >= M) return;
    const int nbase = blockIdx.x * 32;
    const int lrow  = lane & 15;
    const int khi   = lane >> 4;
    const int row   = mbase + lrow;
    const int n0    = nbase + lrow;
    const int n1    = n0 + 16;

    v8f c0 = {}; v8f c1 = {};
    wmma_tile(A, Bt, K, row, n0, n1, khi, c0, c1);

    const float bv0 = bias[n0], bv1 = bias[n1];
    #pragma unroll
    for (int r = 0; r < 8; ++r) {
        int orow = mbase + 8 * khi + r;
        float v0 = c0[r] + bv0;
        float v1 = c1[r] + bv1;
        size_t o0 = (size_t)orow * N + n0;
        size_t o1 = (size_t)orow * N + n1;
        Out0[o0] = v0; Out0[o1] = v1;
        Out1[o0] = v0; Out1[o1] = v1;
    }
}

// ---------------------------------------------------------------------------
// gcn_norm: deg (with self loop) -> dinv -> per-edge norm
// ---------------------------------------------------------------------------
__global__ void deg_init(float* __restrict__ deg, int n) {
    int i = blockIdx.x * blockDim.x + threadIdx.x;
    if (i < n) deg[i] = 1.0f;                      // self loop contributes 1
}

__global__ void deg_accum(const int* __restrict__ dst, float* __restrict__ deg, int E) {
    int e = blockIdx.x * blockDim.x + threadIdx.x;
    if (e < E) atomicAdd(&deg[dst[e]], 1.0f);
}

__global__ void dinv_kernel(const float* __restrict__ deg, float* __restrict__ dinv, int n) {
    int i = blockIdx.x * blockDim.x + threadIdx.x;
    if (i < n) dinv[i] = rsqrtf(deg[i]);           // deg >= 1 always
}

__global__ void norm_kernel(const int* __restrict__ src, const int* __restrict__ dst,
                            const float* __restrict__ dinv, float* __restrict__ norm, int E) {
    int e = blockIdx.x * blockDim.x + threadIdx.x;
    if (e < E) norm[e] = dinv[src[e]] * dinv[dst[e]];
}

// ---------------------------------------------------------------------------
// APPNP step: out = alpha*h0 + (1-alpha)*(selfnorm*h + sum_edges norm*h[src])
// prop_init writes self-loop + teleport terms (full overwrite), then
// prop_edges atomically accumulates edge terms. Working set (~115 MB:
// edges + norm + 3 feature buffers) fits in the 192 MB L2.
// ---------------------------------------------------------------------------
__global__ void prop_init(const float* __restrict__ h, const float* __restrict__ h0,
                          const float* __restrict__ dinv, float* __restrict__ out, long total) {
    long i = (long)blockIdx.x * blockDim.x + threadIdx.x;
    if (i >= total) return;
    int node = (int)(i >> 6);                      // N_CLS == 64
    float di = dinv[node];
    out[i] = ALPHA * h0[i] + (1.0f - ALPHA) * di * di * h[i];
}

__global__ void prop_edges(const int* __restrict__ src, const int* __restrict__ dst,
                           const float* __restrict__ norm,
                           const float* __restrict__ h, float* __restrict__ out, int E) {
    int e = blockIdx.x * (blockDim.x >> 5) + (threadIdx.x >> 5);   // one wave per edge
    if (e >= E) return;
    int lane = threadIdx.x & 31;
    int s = src[e], d = dst[e];
    float w = (1.0f - ALPHA) * norm[e];
    float2 v = *(const float2*)(h + (size_t)s * N_CLS + lane * 2);
    atomicAdd(&out[(size_t)d * N_CLS + lane * 2 + 0], w * v.x);
    atomicAdd(&out[(size_t)d * N_CLS + lane * 2 + 1], w * v.y);
}

// ---------------------------------------------------------------------------
// log_softmax over 64 classes: one wave per node, 2 elements per lane
// ---------------------------------------------------------------------------
__global__ void logsoftmax_kernel(const float* __restrict__ h, float* __restrict__ out, int n) {
    int node = blockIdx.x * (blockDim.x >> 5) + (threadIdx.x >> 5);
    if (node >= n) return;
    int lane = threadIdx.x & 31;
    float2 v = *(const float2*)(h + (size_t)node * N_CLS + lane * 2);
    float m = fmaxf(v.x, v.y);
    #pragma unroll
    for (int o = 16; o > 0; o >>= 1) m = fmaxf(m, __shfl_xor(m, o, 32));
    float s = __expf(v.x - m) + __expf(v.y - m);
    #pragma unroll
    for (int o = 16; o > 0; o >>= 1) s += __shfl_xor(s, o, 32);
    float lg = m + __logf(s);
    out[(size_t)node * N_CLS + lane * 2 + 0] = v.x - lg;
    out[(size_t)node * N_CLS + lane * 2 + 1] = v.y - lg;
}

// ---------------------------------------------------------------------------
extern "C" void kernel_launch(void* const* d_in, const int* in_sizes, int n_in,
                              void* d_out, int out_size, void* d_ws, size_t ws_size,
                              hipStream_t stream) {
    const float* x   = (const float*)d_in[0];
    const int*   ei  = (const int*)d_in[1];
    const int*   src = ei;
    const int*   dst = ei + N_EDGES;
    const float* W1  = (const float*)d_in[2];
    const float* b1  = (const float*)d_in[3];
    const float* W2  = (const float*)d_in[4];
    const float* b2  = (const float*)d_in[5];
    float*       out = (float*)d_out;

    // Workspace carve-out (256B aligned)
    char* ws = (char*)d_ws;
    size_t off = 0;
    auto carve = [&](size_t bytes) -> char* {
        char* p = ws + off;
        off += (bytes + 255) & ~(size_t)255;
        return p;
    };
    _Float16* xh   = (_Float16*)carve((size_t)N_NODES * F_IN * 2);   // x in f16
    _Float16* w1t  = (_Float16*)carve((size_t)HID * F_IN * 2);       // W1^T f16 [HID, F_IN]
    _Float16* w2t  = (_Float16*)carve((size_t)N_CLS * HID * 2);      // W2^T f16 [N_CLS, HID]
    _Float16* h1h  = (_Float16*)carve((size_t)N_NODES * HID * 2);    // relu(xW1+b1) f16
    float*    h0   = (float*)carve((size_t)N_NODES * N_CLS * 4);
    float*    hA   = (float*)carve((size_t)N_NODES * N_CLS * 4);
    float*    hB   = (float*)carve((size_t)N_NODES * N_CLS * 4);
    float*    deg  = (float*)carve((size_t)N_NODES * 4);
    float*    dinv = (float*)carve((size_t)N_NODES * 4);
    float*    nrm  = (float*)carve((size_t)N_EDGES * 4);

    const int BT = 256;

    // 1) precision conversion + weight transposes
    {
        long n4 = (long)N_NODES * F_IN / 4;
        cvt_f32_f16_v4<<<(int)((n4 + BT - 1) / BT), BT, 0, stream>>>(x, xh, n4);
        cvt_transpose_f16<<<(F_IN * HID + BT - 1) / BT, BT, 0, stream>>>(W1, w1t, F_IN, HID);
        cvt_transpose_f16<<<(HID * N_CLS + BT - 1) / BT, BT, 0, stream>>>(W2, w2t, HID, N_CLS);
    }

    // 2) GEMM1: h1 = relu(x @ W1 + b1)   [100000 x 512]
    {
        dim3 grid(HID / 32, (N_NODES + 63) / 64);
        gemm_wmma_relu_f16<<<grid, 128, 0, stream>>>(xh, w1t, b1, h1h, N_NODES, HID, F_IN);
    }

    // 3) GEMM2: logits = h1 @ W2 + b2 -> h0 and hA   [100000 x 64]
    {
        dim3 grid(N_CLS / 32, (N_NODES + 63) / 64);
        gemm_wmma_f32x2<<<grid, 128, 0, stream>>>(h1h, w2t, b2, h0, hA, N_NODES, N_CLS, HID);
    }

    // 4) gcn_norm
    deg_init<<<(N_NODES + BT - 1) / BT, BT, 0, stream>>>(deg, N_NODES);
    deg_accum<<<(N_EDGES + BT - 1) / BT, BT, 0, stream>>>(dst, deg, N_EDGES);
    dinv_kernel<<<(N_NODES + BT - 1) / BT, BT, 0, stream>>>(deg, dinv, N_NODES);
    norm_kernel<<<(N_EDGES + BT - 1) / BT, BT, 0, stream>>>(src, dst, dinv, nrm, N_EDGES);

    // 5) APPNP propagation, K steps (ping-pong hA/hB; working set lives in L2)
    float* cur = hA;
    float* nxt = hB;
    const long total = (long)N_NODES * N_CLS;
    const int wavesPerBlock = BT / 32;
    for (int k = 0; k < K_STEPS; ++k) {
        prop_init<<<(int)((total + BT - 1) / BT), BT, 0, stream>>>(cur, h0, dinv, nxt, total);
        prop_edges<<<(N_EDGES + wavesPerBlock - 1) / wavesPerBlock, BT, 0, stream>>>(
            src, dst, nrm, cur, nxt, N_EDGES);
        float* t = cur; cur = nxt; nxt = t;
    }

    // 6) log_softmax -> d_out
    logsoftmax_kernel<<<(N_NODES + wavesPerBlock - 1) / wavesPerBlock, BT, 0, stream>>>(
        cur, out, N_NODES);
}
